// BidirectionalAttention_76845554860501
// MI455X (gfx1250) — matrix operand order repaired and checked by reading, not compiled, not run on hardware
//
#include <hip/hip_runtime.h>
#include <math.h>

// ---------------- types ----------------
typedef __bf16 bf16_t;
typedef __attribute__((ext_vector_type(16))) __bf16 v16bf;
typedef __attribute__((ext_vector_type(8)))  __bf16 v8bf;
typedef __attribute__((ext_vector_type(8)))  float  v8f;

union Frag16 { v16bf v; v8bf h[2]; };

// ---------------- problem constants ----------------
constexpr int    Dm = 768;
constexpr int    Lm = 1024;
constexpr int    NB = 8;
constexpr float  INV_SCALE = 0.036084391824351615f;   // 1/sqrt(768)
constexpr size_t S1 = (size_t)NB * Lm * Dm * 2;       // bf16 (1024x768) x 8 batches
constexpr size_t S2 = (size_t)NB * Lm * Lm * 4;       // f32 scores
constexpr size_t S3 = (size_t)NB * Lm * Lm * 2;       // bf16 attn
constexpr size_t OUT_HALF = (size_t)NB * Lm * (2 * Dm); // elements per output tensor

// ---------------- fast tanh (gfx1250 V_TANH_F32 when available) ----------------
__device__ __forceinline__ float fast_tanh(float x) {
#if __has_builtin(__builtin_amdgcn_tanhf)
    return __builtin_amdgcn_tanhf(x);
#else
    // branchless: tanh(x) = 1 - 2/(exp(2x)+1); exp overflow/underflow saturate correctly
    return 1.0f - 2.0f / (__expf(2.0f * x) + 1.0f);
#endif
}

// ---------------- WMMA fragment loaders (pure bf16) ----------------
// A operand (16x32 bf16): lane<16 holds K={0..7,16..23}, lane>=16 K={8..15,24..31}
__device__ __forceinline__ v16bf load_frag_a_bf16(const bf16_t* tile, int ld) {
    int lane = threadIdx.x & 31;
    int r = lane & 15, hi = lane >> 4;
    const bf16_t* p = tile + (size_t)r * ld + hi * 8;
    Frag16 f;
    f.h[0] = *(const v8bf*)(p);
    f.h[1] = *(const v8bf*)(p + 16);
    return f.v;
}
// B operand (32x16 bf16): lane<16 holds K={0..7,16..23}, lane>=16 K={8..15,24..31}
__device__ __forceinline__ v16bf load_frag_b_bf16(const bf16_t* tile, int ld) {
    int lane = threadIdx.x & 31;
    int nn = lane & 15, kh = lane >> 4;
    const bf16_t* p = tile + (size_t)nn * ld + kh * 8;
    Frag16 f;
    f.h[0] = *(const v8bf*)(p);
    f.h[1] = *(const v8bf*)(p + 16);
    return f.v;
}

#define WMMA_BF16(A, B, C) \
    __builtin_amdgcn_wmma_f32_16x16x32_bf16(false, (A), false, (B), (short)0, (C), false, false)

// ---------------- kernel 1: copy to output halves + bf16 row-major + bf16 transposes ----
// grid (32 l-tiles, 24 d-tiles, 16 = n*2+side), block 256
__global__ void __launch_bounds__(256)
prep_kernel(const float* __restrict__ lhs, const float* __restrict__ rhs,
            float* __restrict__ out,
            bf16_t* __restrict__ lhs_b, bf16_t* __restrict__ rhs_b,
            bf16_t* __restrict__ lhsT,  bf16_t* __restrict__ rhsT) {
    int side = blockIdx.z & 1;
    int n    = blockIdx.z >> 1;
    const float* X  = (side ? rhs : lhs) + (size_t)n * Lm * Dm;
    float*       O  = out + (side ? OUT_HALF : 0) + (size_t)n * Lm * (2 * Dm);
    bf16_t*      Xb = (side ? rhs_b : lhs_b) + (size_t)n * Lm * Dm;
    bf16_t*      T  = (side ? rhsT : lhsT) + (size_t)n * Dm * Lm;
    int l0 = blockIdx.x * 32, d0 = blockIdx.y * 32;
    __shared__ float tile[32][33];
    int tx = threadIdx.x & 31, tyb = threadIdx.x >> 5;
#pragma unroll
    for (int p = 0; p < 4; ++p) {
        int l = l0 + tyb + p * 8;
        float v = X[(size_t)l * Dm + d0 + tx];
        tile[tyb + p * 8][tx] = v;
        O[(size_t)l * (2 * Dm) + d0 + tx] = v;        // concat first half = raw input
        Xb[(size_t)l * Dm + d0 + tx] = (bf16_t)v;     // bf16 row-major copy
    }
    __syncthreads();
#pragma unroll
    for (int p = 0; p < 4; ++p) {
        int d = d0 + tyb + p * 8;
        T[(size_t)d * Lm + l0 + tx] = (bf16_t)tile[tx][tyb + p * 8];
    }
}

// ---------------- kernel 1b: W fp32 -> bf16 ----------------
// grid (576, 2), block 256, 4 elems/thread
__global__ void __launch_bounds__(256)
wconv_kernel(const float* __restrict__ W_lhs, const float* __restrict__ W_rhs,
             bf16_t* __restrict__ Wb) {
    int side = blockIdx.y;
    const float* W = side ? W_rhs : W_lhs;
    bf16_t* Ob = Wb + (size_t)side * Dm * Dm;
    size_t base = ((size_t)blockIdx.x * 256 + threadIdx.x) * 4;
#pragma unroll
    for (int i = 0; i < 4; ++i) Ob[base + i] = (bf16_t)W[base + i];
}

// ---------------- kernel 2: l/r = tanh(Xb @ Wb^T), bf16 out ----------------
// grid (3, 64, 2), block 256 (8 waves, each 64x64 tile)
__global__ void __launch_bounds__(256)
linear_tanh_kernel(const bf16_t* __restrict__ lhs_b, const bf16_t* __restrict__ rhs_b,
                   const bf16_t* __restrict__ Wb,
                   bf16_t* __restrict__ l_out, bf16_t* __restrict__ r_out) {
    int side = blockIdx.z;
    const bf16_t* X = side ? rhs_b : lhs_b;               // (8192, 768) bf16
    const bf16_t* W = Wb + (size_t)side * Dm * Dm;        // (768, 768) bf16, B[k][n]=W[n][k]
    bf16_t*       O = side ? r_out : l_out;               // (8192, 768) bf16
    int wave = threadIdx.x >> 5;
    int wm = wave >> 2, wn = wave & 3;
    int row0 = blockIdx.y * 128 + wm * 64;
    int col0 = blockIdx.x * 256 + wn * 64;
    v8f acc[4][4];
#pragma unroll
    for (int i = 0; i < 4; ++i)
#pragma unroll
        for (int j = 0; j < 4; ++j)
            acc[i][j] = (v8f){0.f,0.f,0.f,0.f,0.f,0.f,0.f,0.f};

    for (int k0 = 0; k0 < Dm; k0 += 32) {
        v16bf af[4], bfv[4];
#pragma unroll
        for (int i = 0; i < 4; ++i)
            af[i] = load_frag_a_bf16(X + (size_t)(row0 + 16 * i) * Dm + k0, Dm);
#pragma unroll
        for (int j = 0; j < 4; ++j)
            bfv[j] = load_frag_b_bf16(W + (size_t)(col0 + 16 * j) * Dm + k0, Dm);
#pragma unroll
        for (int i = 0; i < 4; ++i)
#pragma unroll
            for (int j = 0; j < 4; ++j)
                acc[i][j] = WMMA_BF16(af[i], bfv[j], acc[i][j]);
    }
    int lane = threadIdx.x & 31, nn = lane & 15, hi = lane >> 4;
#pragma unroll
    for (int i = 0; i < 4; ++i)
#pragma unroll
        for (int j = 0; j < 4; ++j)
#pragma unroll
            for (int e = 0; e < 8; ++e) {
                int row = row0 + 16 * i + hi * 8 + e;
                int col = col0 + 16 * j + nn;
                O[(size_t)row * Dm + col] = (bf16_t)fast_tanh(acc[i][j][e]);
            }
}

// ---------------- kernel 3: scores = l @ r^T, f32 out ----------------
// grid (4, 8, 8 batches), block 256
__global__ void __launch_bounds__(256)
scores_kernel(const bf16_t* __restrict__ l_b, const bf16_t* __restrict__ r_b,
              float* __restrict__ scores) {
    int n = blockIdx.z;
    const bf16_t* A = l_b + (size_t)n * Lm * Dm;
    const bf16_t* B = r_b + (size_t)n * Lm * Dm;   // B[k][col] = r[col][k]
    float*        C = scores + (size_t)n * Lm * Lm;
    int wave = threadIdx.x >> 5;
    int wm = wave >> 2, wn = wave & 3;
    int row0 = blockIdx.y * 128 + wm * 64;
    int col0 = blockIdx.x * 256 + wn * 64;
    v8f acc[4][4];
#pragma unroll
    for (int i = 0; i < 4; ++i)
#pragma unroll
        for (int j = 0; j < 4; ++j)
            acc[i][j] = (v8f){0.f,0.f,0.f,0.f,0.f,0.f,0.f,0.f};

    for (int k0 = 0; k0 < Dm; k0 += 32) {
        v16bf af[4], bfv[4];
#pragma unroll
        for (int i = 0; i < 4; ++i)
            af[i] = load_frag_a_bf16(A + (size_t)(row0 + 16 * i) * Dm + k0, Dm);
#pragma unroll
        for (int j = 0; j < 4; ++j)
            bfv[j] = load_frag_b_bf16(B + (size_t)(col0 + 16 * j) * Dm + k0, Dm);
#pragma unroll
        for (int i = 0; i < 4; ++i)
#pragma unroll
            for (int j = 0; j < 4; ++j)
                acc[i][j] = WMMA_BF16(af[i], bfv[j], acc[i][j]);
    }
    int lane = threadIdx.x & 31, nn = lane & 15, hi = lane >> 4;
#pragma unroll
    for (int i = 0; i < 4; ++i)
#pragma unroll
        for (int j = 0; j < 4; ++j)
#pragma unroll
            for (int e = 0; e < 8; ++e) {
                int row = row0 + 16 * i + hi * 8 + e;
                int col = col0 + 16 * j + nn;
                C[(size_t)row * Lm + col] = acc[i][j][e];
            }
}

// ---------------- kernel 4: row softmax -> bf16 attn_lhs ----------------
// grid 8192 rows, block 256
__global__ void __launch_bounds__(256)
softmax_row_kernel(const float* __restrict__ scores, bf16_t* __restrict__ attn) {
    size_t row = blockIdx.x;
    const float* S = scores + row * Lm;
    bf16_t*      A = attn + row * Lm;
    int tid = threadIdx.x, wave = tid >> 5, lane = tid & 31;
    __shared__ float red[8];
    __shared__ float bc;
    float v[4];
    float m = -3.4e38f;
#pragma unroll
    for (int i = 0; i < 4; ++i) { v[i] = S[tid + i * 256]; m = fmaxf(m, v[i]); }
#pragma unroll
    for (int o = 16; o > 0; o >>= 1) m = fmaxf(m, __shfl_xor(m, o, 32));
    if (lane == 0) red[wave] = m;
    __syncthreads();
    if (tid == 0) { float x = red[0]; for (int i = 1; i < 8; ++i) x = fmaxf(x, red[i]); bc = x; }
    __syncthreads();
    m = bc;
    float s = 0.f;
#pragma unroll
    for (int i = 0; i < 4; ++i) { v[i] = __expf(v[i] - m); s += v[i]; }
#pragma unroll
    for (int o = 16; o > 0; o >>= 1) s += __shfl_xor(s, o, 32);
    __syncthreads();
    if (lane == 0) red[wave] = s;
    __syncthreads();
    if (tid == 0) { float x = 0.f; for (int i = 0; i < 8; ++i) x += red[i]; bc = x; }
    __syncthreads();
    float inv = 1.0f / bc;
#pragma unroll
    for (int i = 0; i < 4; ++i) A[tid + i * 256] = (bf16_t)(v[i] * inv);
}

// ---------------- kernel 5: column softmax (scaled) -> bf16 attn_rhs^T ----------------
// grid (64 col-groups, 8 batches), block 256 (tx=col 0..15, ty=row-slice 0..15)
__global__ void __launch_bounds__(256)
softmax_col_kernel(const float* __restrict__ scores, bf16_t* __restrict__ attn_rhsT) {
    int n  = blockIdx.y;
    int tx = threadIdx.x & 15;
    int ty = threadIdx.x >> 4;
    int c  = blockIdx.x * 16 + tx;
    const float* S = scores + (size_t)n * Lm * Lm;
    float vals[64];
    float m = -3.4e38f;
#pragma unroll
    for (int i = 0; i < 64; ++i) {
        int l = ty + i * 16;
        float x = S[(size_t)l * Lm + c] * INV_SCALE;
        vals[i] = x;
        m = fmaxf(m, x);
    }
    __shared__ float red[16][17];
    red[ty][tx] = m;
    __syncthreads();
#pragma unroll
    for (int s = 8; s > 0; s >>= 1) {
        if (ty < s) red[ty][tx] = fmaxf(red[ty][tx], red[ty + s][tx]);
        __syncthreads();
    }
    float cmax = red[0][tx];
    __syncthreads();
    float sum = 0.f;
#pragma unroll
    for (int i = 0; i < 64; ++i) { vals[i] = __expf(vals[i] - cmax); sum += vals[i]; }
    red[ty][tx] = sum;
    __syncthreads();
#pragma unroll
    for (int s = 8; s > 0; s >>= 1) {
        if (ty < s) red[ty][tx] += red[ty + s][tx];
        __syncthreads();
    }
    float inv = 1.0f / red[0][tx];
    bf16_t* T = attn_rhsT + (size_t)n * Lm * Lm + (size_t)c * Lm;
#pragma unroll
    for (int i = 0; i < 64; ++i) T[ty + i * 16] = (bf16_t)(vals[i] * inv);
}

// ---------------- kernel 6: out = attn @ X into strided concat output ----------------
// grid (3, 8, 16 = n*2+side), block 256
__global__ void __launch_bounds__(256)
attn_out_kernel(const bf16_t* __restrict__ attn_lhs, const bf16_t* __restrict__ attn_rhsT,
                const bf16_t* __restrict__ rhsT, const bf16_t* __restrict__ lhsT,
                float* __restrict__ out) {
    int side = blockIdx.z & 1;
    int n    = blockIdx.z >> 1;
    const bf16_t* A = (side ? attn_rhsT : attn_lhs) + (size_t)n * Lm * Lm;  // (1024,1024)
    const bf16_t* B = (side ? lhsT : rhsT) + (size_t)n * Dm * Lm;           // (768,1024): B[k][d]=X[k][d]
    float*        O = out + (side ? OUT_HALF : 0) + (size_t)n * Lm * (2 * Dm);
    int wave = threadIdx.x >> 5;
    int wm = wave >> 2, wn = wave & 3;
    int row0 = blockIdx.y * 128 + wm * 64;
    int col0 = blockIdx.x * 256 + wn * 64;
    v8f acc[4][4];
#pragma unroll
    for (int i = 0; i < 4; ++i)
#pragma unroll
        for (int j = 0; j < 4; ++j)
            acc[i][j] = (v8f){0.f,0.f,0.f,0.f,0.f,0.f,0.f,0.f};

    for (int k0 = 0; k0 < Lm; k0 += 32) {
        v16bf af[4], bfv[4];
#pragma unroll
        for (int i = 0; i < 4; ++i)
            af[i] = load_frag_a_bf16(A + (size_t)(row0 + 16 * i) * Lm + k0, Lm);
#pragma unroll
        for (int j = 0; j < 4; ++j)
            bfv[j] = load_frag_b_bf16(B + (size_t)(col0 + 16 * j) * Lm + k0, Lm);
#pragma unroll
        for (int i = 0; i < 4; ++i)
#pragma unroll
            for (int j = 0; j < 4; ++j)
                acc[i][j] = WMMA_BF16(af[i], bfv[j], acc[i][j]);
    }
    int lane = threadIdx.x & 31, nn = lane & 15, hi = lane >> 4;
#pragma unroll
    for (int i = 0; i < 4; ++i)
#pragma unroll
        for (int j = 0; j < 4; ++j)
#pragma unroll
            for (int e = 0; e < 8; ++e) {
                int row = row0 + 16 * i + hi * 8 + e;
                int col = col0 + 16 * j + nn;
                O[(size_t)row * (2 * Dm) + Dm + col] = acc[i][j][e];  // concat second half
            }
}

// ---------------- host ----------------
extern "C" void kernel_launch(void* const* d_in, const int* in_sizes, int n_in,
                              void* d_out, int out_size, void* d_ws, size_t ws_size,
                              hipStream_t stream) {
    const float* lhs   = (const float*)d_in[0];
    const float* rhs   = (const float*)d_in[1];
    const float* W_lhs = (const float*)d_in[2];
    const float* W_rhs = (const float*)d_in[3];
    float* out = (float*)d_out;

    char* ws = (char*)d_ws;
    bf16_t* l_b   = (bf16_t*)(ws);                          // tanh(lhs@W^T) bf16 (8,1024,768)
    bf16_t* r_b   = (bf16_t*)(ws + S1);                     // tanh(rhs@W^T) bf16
    bf16_t* lhsT  = (bf16_t*)(ws + 2 * S1);                 // lhs^T bf16 (8,768,1024)
    bf16_t* rhsT  = (bf16_t*)(ws + 3 * S1);                 // rhs^T bf16
    bf16_t* lhs_b = (bf16_t*)(ws + 4 * S1);                 // lhs bf16 row-major
    bf16_t* rhs_b = (bf16_t*)(ws + 5 * S1);                 // rhs bf16 row-major
    float*  sc    = (float*) (ws + 6 * S1);                 // scores f32 (8,1024,1024)
    bf16_t* a_l   = (bf16_t*)(ws + 6 * S1 + S2);            // attn_lhs bf16
    bf16_t* a_rT  = (bf16_t*)(ws + 6 * S1 + S2 + S3);       // attn_rhs^T bf16
    bf16_t* W_b   = (bf16_t*)(ws + 6 * S1 + S2 + 2 * S3);   // both W bf16 (2,768,768)

    prep_kernel<<<dim3(32, 24, 16), 256, 0, stream>>>(lhs, rhs, out, lhs_b, rhs_b, lhsT, rhsT);
    wconv_kernel<<<dim3(576, 2), 256, 0, stream>>>(W_lhs, W_rhs, W_b);
    linear_tanh_kernel<<<dim3(3, 64, 2), 256, 0, stream>>>(lhs_b, rhs_b, W_b, l_b, r_b);
    scores_kernel<<<dim3(4, 8, 8), 256, 0, stream>>>(l_b, r_b, sc);
    softmax_row_kernel<<<dim3(8192), 256, 0, stream>>>(sc, a_l);
    softmax_col_kernel<<<dim3(64, 8), 256, 0, stream>>>(sc, a_rT);
    attn_out_kernel<<<dim3(3, 8, 16), 256, 0, stream>>>(a_l, a_rT, rhsT, lhsT, out);
}
